// HierarchicalMixtureOfExperts_34823594836462
// MI455X (gfx1250) — matrix operand order repaired and checked by reading, not compiled, not run on hardware
//
#include <hip/hip_runtime.h>
#include <math.h>

typedef __attribute__((ext_vector_type(16))) _Float16 v16h;
typedef __attribute__((ext_vector_type(8)))  _Float16 h8;
typedef __attribute__((ext_vector_type(4)))  _Float16 h4;
typedef __attribute__((ext_vector_type(2)))  _Float16 h2;
typedef __attribute__((ext_vector_type(8)))  float    v8f;

__device__ __forceinline__ float gelu_exact(float x) {
    return 0.5f * x * (1.0f + erff(x * 0.70710678118654752f));
}

// ---------------------------------------------------------------------------
// Generic f32 -> f16 conversion (weights, activations)
// ---------------------------------------------------------------------------
__global__ void cvt_f32_f16_kernel(const float* __restrict__ src,
                                   _Float16* __restrict__ dst, long n) {
    long i = blockIdx.x * (long)blockDim.x + threadIdx.x;
    long stride = (long)gridDim.x * blockDim.x;
    for (; i < n; i += stride) dst[i] = (_Float16)src[i];
}

// ---------------------------------------------------------------------------
// pos = x + sinusoidal PE, stored f16
// ---------------------------------------------------------------------------
__global__ void add_pe_kernel(const float* __restrict__ x,
                              _Float16* __restrict__ pos16,
                              long n, int S, int D) {
    long i = blockIdx.x * (long)blockDim.x + threadIdx.x;
    long stride = (long)gridDim.x * blockDim.x;
    const float negln = -9.210340371976184f; // -ln(10000)
    for (; i < n; i += stride) {
        int d = (int)(i % D);
        long tok = i / D;
        int s = (int)(tok % S);
        float div = expf((float)(d & ~1) * (negln / (float)D));
        float ang = (float)s * div;
        float pe = (d & 1) ? cosf(ang) : sinf(ang);
        pos16[i] = (_Float16)(x[i] + pe);
    }
}

// ---------------------------------------------------------------------------
// Tiled WMMA GEMM: C[M,N] = A[M,K] (f16) * B[K,N] (f16) + bias epilogue.
// Block = 256 threads = 8 waves; block tile 128x64; wave tile 32x32 (2x2 WMMA);
// BK = 64 (two 16x16x32 sub-steps -> 8 WMMA per wave per slab).
// Double-buffered LDS, software-pipelined: next-slab global loads issue before
// current-slab WMMAs; one barrier per slab.
// M%128==0, N%64==0, K%64==0 (guaranteed by problem dims).
// MODE 0: Ch = f16(gelu(acc + bias))
// MODE 1: Cf = acc + bias
// MODE 2: Cf += rowscale[row*rs_stride] * (acc + bias)
// ---------------------------------------------------------------------------
template <int MODE>
__global__ __launch_bounds__(256) void gemm_wmma_f16(
    const _Float16* __restrict__ A, const _Float16* __restrict__ B,
    const float* __restrict__ bias,
    const float* __restrict__ rowscale, int rs_stride,
    _Float16* __restrict__ Ch, float* __restrict__ Cf,
    int M, int N, int K) {
    __shared__ _Float16 As[2][128][72];   // [buf][row][k], 144B rows
    __shared__ _Float16 Bst[2][64][72];   // [buf][col][k] (transposed)

    const int tid  = threadIdx.x;
    const int lane = tid & 31;
    const int wave = tid >> 5;
    const int wm   = wave & 3;       // 0..3  -> 32-row strip
    const int wn   = wave >> 2;      // 0..1  -> 32-col strip
    const int hi   = (lane >> 4) & 1;
    const int ln16 = lane & 15;
    const int row0 = blockIdx.y * 128;
    const int col0 = blockIdx.x * 64;

    // per-thread staging coordinates (compile-time unrollable)
    int ar_r[4], ar_c[4];
    #pragma unroll
    for (int i = 0; i < 4; ++i) {
        int linear = (i * 256 + tid) * 8;
        ar_r[i] = linear >> 6;
        ar_c[i] = linear & 63;
    }
    int bk[2], bc[2];
    #pragma unroll
    for (int i = 0; i < 2; ++i) {
        int sub = i * 256 + tid;       // 512 sub-blocks of 2(k) x 4(col)
        bk[i] = (sub >> 4) * 2;
        bc[i] = (sub & 15) * 4;
    }

    h8 areg[4];
    h4 b0reg[2], b1reg[2];

    // phase 1: issue all global loads for slab k0 (kept in flight together)
    auto load_regs = [&](int k0) {
        #pragma unroll
        for (int i = 0; i < 4; ++i)
            areg[i] = *(const h8*)&A[(size_t)(row0 + ar_r[i]) * K + k0 + ar_c[i]];
        #pragma unroll
        for (int i = 0; i < 2; ++i) {
            b0reg[i] = *(const h4*)&B[(size_t)(k0 + bk[i])     * N + col0 + bc[i]];
            b1reg[i] = *(const h4*)&B[(size_t)(k0 + bk[i] + 1) * N + col0 + bc[i]];
        }
    };
    // phase 2: commit staged registers to LDS buffer `buf`
    auto store_lds = [&](int buf) {
        #pragma unroll
        for (int i = 0; i < 4; ++i)
            *(h8*)&As[buf][ar_r[i]][ar_c[i]] = areg[i];
        #pragma unroll
        for (int i = 0; i < 2; ++i)
            #pragma unroll
            for (int j = 0; j < 4; ++j) {
                h2 pk = {b0reg[i][j], b1reg[i][j]};
                *(h2*)&Bst[buf][bc[i] + j][bk[i]] = pk;
            }
    };

    v8f acc[2][2] = {};

    auto compute = [&](int buf) {
        #pragma unroll
        for (int ks = 0; ks < 64; ks += 32) {
            v16h afrag[2], bfrag[2];
            #pragma unroll
            for (int mt = 0; mt < 2; ++mt) {
                int row = wm * 32 + mt * 16 + ln16;
                h8 lo = *(const h8*)&As[buf][row][ks + hi * 8];
                h8 hh = *(const h8*)&As[buf][row][ks + 16 + hi * 8];
                afrag[mt] = __builtin_shufflevector(lo, hh,
                    0,1,2,3,4,5,6,7,8,9,10,11,12,13,14,15);
            }
            #pragma unroll
            for (int nt = 0; nt < 2; ++nt) {
                int col = wn * 32 + nt * 16 + ln16;
                h8 lo = *(const h8*)&Bst[buf][col][ks + hi * 8];
                h8 hh = *(const h8*)&Bst[buf][col][ks + 16 + hi * 8];
                bfrag[nt] = __builtin_shufflevector(lo, hh,
                    0,1,2,3,4,5,6,7,8,9,10,11,12,13,14,15);
            }
            #pragma unroll
            for (int mt = 0; mt < 2; ++mt)
                #pragma unroll
                for (int nt = 0; nt < 2; ++nt)
                    acc[mt][nt] = __builtin_amdgcn_wmma_f32_16x16x32_f16(
                        false, afrag[mt], false, bfrag[nt],
                        (short)0, acc[mt][nt], false, false);
        }
    };

    // prologue: stage slab 0
    load_regs(0);
    store_lds(0);
    __syncthreads();

    int cur = 0;
    for (int k0 = 0; k0 < K; k0 += 64) {
        if (k0 + 64 < K) {
            load_regs(k0 + 64);   // next slab's global loads go in flight
            __builtin_prefetch(&A[(size_t)(row0 + (tid >> 1)) * K + k0 + 128], 0, 1);
            compute(cur);         // WMMA on current slab overlaps the loads
            store_lds(cur ^ 1);   // other waves only read buf `cur` -> no hazard
        } else {
            compute(cur);
        }
        __syncthreads();          // single barrier per slab
        cur ^= 1;
    }

    // ---- epilogue; C/D layout: lane<16 -> M=v, lane>=16 -> M=v+8, N=lane&15
    #pragma unroll
    for (int mt = 0; mt < 2; ++mt) {
        #pragma unroll
        for (int nt = 0; nt < 2; ++nt) {
            int gm0 = row0 + wm * 32 + mt * 16 + hi * 8;
            int gn  = col0 + wn * 32 + nt * 16 + ln16;
            float bcol = bias ? bias[gn] : 0.0f;
            #pragma unroll
            for (int v = 0; v < 8; ++v) {
                int gmr = gm0 + v;
                float val = acc[mt][nt][v] + bcol;
                size_t idx = (size_t)gmr * N + gn;
                if (MODE == 0) {
                    Ch[idx] = (_Float16)gelu_exact(val);
                } else if (MODE == 1) {
                    Cf[idx] = val;
                } else {
                    Cf[idx] += rowscale[(size_t)gmr * rs_stride] * val;
                }
            }
        }
    }
}

// ---------------------------------------------------------------------------
// Router: logits = (rh @ Wr2 + br2)/(temp+1e-6); softmax; top-2 sparse weights.
// One wave (32 lanes) per token; E=8.
// ---------------------------------------------------------------------------
__global__ __launch_bounds__(256) void router_kernel(
    const _Float16* __restrict__ rh, const float* __restrict__ Wr2,
    const float* __restrict__ br2, const float* __restrict__ temp,
    float* __restrict__ logits_out, float* __restrict__ tokw,
    int NT, int Hr) {
    int lane = threadIdx.x & 31;
    int wave = threadIdx.x >> 5;
    int t = blockIdx.x * 8 + wave;
    if (t >= NT) return;
    const _Float16* r = rh + (size_t)t * Hr;
    float acc[8] = {};
    for (int k = lane; k < Hr; k += 32) {
        float rv = (float)r[k];
        const float* w = Wr2 + (size_t)k * 8;
        #pragma unroll
        for (int e = 0; e < 8; ++e) acc[e] = fmaf(rv, w[e], acc[e]);
    }
    #pragma unroll
    for (int e = 0; e < 8; ++e)
        #pragma unroll
        for (int off = 16; off > 0; off >>= 1)
            acc[e] += __shfl_xor(acc[e], off, 32);
    if (lane == 0) {
        float inv_t = 1.0f / (temp[0] + 1e-6f);
        float lg[8], p[8];
        float m = -3.0e38f;
        #pragma unroll
        for (int e = 0; e < 8; ++e) {
            lg[e] = (acc[e] + br2[e]) * inv_t;
            m = fmaxf(m, lg[e]);
        }
        float s = 0.0f;
        #pragma unroll
        for (int e = 0; e < 8; ++e) { p[e] = expf(lg[e] - m); s += p[e]; }
        float inv_s = 1.0f / s;
        #pragma unroll
        for (int e = 0; e < 8; ++e) p[e] *= inv_s;
        int i1 = 0;
        #pragma unroll
        for (int e = 1; e < 8; ++e) if (p[e] > p[i1]) i1 = e;   // first-on-tie
        int i2 = -1; float b2 = -1.0f;
        #pragma unroll
        for (int e = 0; e < 8; ++e)
            if (e != i1 && p[e] > b2) { b2 = p[e]; i2 = e; }
        #pragma unroll
        for (int e = 0; e < 8; ++e) {
            logits_out[(size_t)t * 8 + e] = lg[e];
            tokw[(size_t)t * 8 + e] = (e == i1) ? p[i1] : ((e == i2) ? p[i2] : 0.0f);
        }
    }
}

// ---------------------------------------------------------------------------
// Deterministic capacity loss: sum_e (sum_t softmax(logits)_te / B)^2
// (mean(load^2)*E == sum(load^2) for E experts). Single block.
// ---------------------------------------------------------------------------
__global__ __launch_bounds__(256) void capacity_kernel(
    const float* __restrict__ logits, float* __restrict__ loss_out,
    int NT, float invB) {
    __shared__ float red[256][8];
    float l[8] = {};
    for (int t = threadIdx.x; t < NT; t += 256) {
        const float* lg = logits + (size_t)t * 8;
        float m = lg[0];
        #pragma unroll
        for (int e = 1; e < 8; ++e) m = fmaxf(m, lg[e]);
        float p[8], s = 0.0f;
        #pragma unroll
        for (int e = 0; e < 8; ++e) { p[e] = expf(lg[e] - m); s += p[e]; }
        float inv = 1.0f / s;
        #pragma unroll
        for (int e = 0; e < 8; ++e) l[e] += p[e] * inv;
    }
    #pragma unroll
    for (int e = 0; e < 8; ++e) red[threadIdx.x][e] = l[e];
    __syncthreads();
    for (int o = 128; o > 0; o >>= 1) {
        if (threadIdx.x < o)
            #pragma unroll
            for (int e = 0; e < 8; ++e)
                red[threadIdx.x][e] += red[threadIdx.x + o][e];
        __syncthreads();
    }
    if (threadIdx.x == 0) {
        float loss = 0.0f;
        #pragma unroll
        for (int e = 0; e < 8; ++e) {
            float el = red[0][e] * invB;
            loss += el * el;
        }
        *loss_out = loss;
    }
}

// ---------------------------------------------------------------------------
// LayerNorm over last dim (D=768). One block per token.
// ---------------------------------------------------------------------------
__global__ __launch_bounds__(256) void ln_kernel(
    const float* __restrict__ c, const float* __restrict__ g,
    const float* __restrict__ b, float* __restrict__ out, int D) {
    __shared__ float s1[256], s2[256];
    int t = blockIdx.x;
    const float* row = c + (size_t)t * D;
    float sum = 0.0f, sq = 0.0f;
    for (int d = threadIdx.x; d < D; d += 256) {
        float v = row[d];
        sum += v; sq += v * v;
    }
    s1[threadIdx.x] = sum; s2[threadIdx.x] = sq;
    __syncthreads();
    for (int o = 128; o > 0; o >>= 1) {
        if (threadIdx.x < o) {
            s1[threadIdx.x] += s1[threadIdx.x + o];
            s2[threadIdx.x] += s2[threadIdx.x + o];
        }
        __syncthreads();
    }
    float mu = s1[0] / (float)D;
    float var = s2[0] / (float)D - mu * mu;
    float inv = rsqrtf(var + 1e-5f);
    for (int d = threadIdx.x; d < D; d += 256)
        out[(size_t)t * D + d] = (row[d] - mu) * inv * g[d] + b[d];
}

// ---------------------------------------------------------------------------
extern "C" void kernel_launch(void* const* d_in, const int* in_sizes, int n_in,
                              void* d_out, int out_size, void* d_ws, size_t ws_size,
                              hipStream_t stream) {
    (void)in_sizes; (void)n_in; (void)out_size; (void)ws_size;
    const int B = 2, S = 2048, D = 768, E = 8, Hr = 3072, He = 3072, Hc = 1536;
    const int NT = B * S;  // 4096 tokens

    const float* x    = (const float*)d_in[0];
    const float* Wr1  = (const float*)d_in[1];
    const float* br1  = (const float*)d_in[2];
    const float* Wr2  = (const float*)d_in[3];
    const float* br2  = (const float*)d_in[4];
    const float* temp = (const float*)d_in[5];
    const float* We1  = (const float*)d_in[6];
    const float* be1  = (const float*)d_in[7];
    const float* We2  = (const float*)d_in[8];
    const float* be2  = (const float*)d_in[9];
    const float* Wc1  = (const float*)d_in[10];
    const float* bc1  = (const float*)d_in[11];
    const float* Wc2  = (const float*)d_in[12];
    const float* bc2  = (const float*)d_in[13];
    const float* lng  = (const float*)d_in[14];
    const float* lnb  = (const float*)d_in[15];

    float* out      = (float*)d_out;                 // [NT, D]
    float* logits   = out + (size_t)NT * D;          // [NT, E]
    float* loss_out = logits + (size_t)NT * E;       // [1]

    // ---- workspace carve-up (256B aligned)
    char* ws = (char*)d_ws;
    size_t off = 0;
    auto alloc = [&](size_t bytes) -> char* {
        char* p = ws + off;
        off += (bytes + 255) & ~(size_t)255;
        return p;
    };
    _Float16* pos16  = (_Float16*)alloc((size_t)NT * D * 2);
    _Float16* rh16   = (_Float16*)alloc((size_t)NT * Hr * 2);
    _Float16* wr1h   = (_Float16*)alloc((size_t)D * Hr * 2);
    _Float16* we1h   = (_Float16*)alloc((size_t)E * D * He * 2);
    _Float16* we2h   = (_Float16*)alloc((size_t)E * He * D * 2);
    _Float16* wc1h   = (_Float16*)alloc((size_t)D * Hc * 2);
    _Float16* wc2h   = (_Float16*)alloc((size_t)Hc * D * 2);
    float*    tokw   = (float*)alloc((size_t)NT * E * 4);
    float*    comb   = (float*)alloc((size_t)NT * D * 4);
    _Float16* comb16 = (_Float16*)alloc((size_t)NT * D * 2);
    _Float16* eh16   = (_Float16*)alloc((size_t)NT * He * 2);
    _Float16* ch16   = (_Float16*)alloc((size_t)NT * Hc * 2);
    float*    cbuf   = (float*)alloc((size_t)NT * D * 4);

    auto cvt = [&](const float* s, _Float16* d, long n) {
        long blocks = (n + 255) / 256;
        int nb = blocks > 4096 ? 4096 : (int)blocks;
        cvt_f32_f16_kernel<<<nb, 256, 0, stream>>>(s, d, n);
    };

    // 1) weights to f16
    cvt(Wr1, wr1h, (long)D * Hr);
    cvt(We1, we1h, (long)E * D * He);
    cvt(We2, we2h, (long)E * He * D);
    cvt(Wc1, wc1h, (long)D * Hc);
    cvt(Wc2, wc2h, (long)Hc * D);

    // 2) pos = x + PE (f16)
    add_pe_kernel<<<4096, 256, 0, stream>>>(x, pos16, (long)NT * D, S, D);

    // 3) router hidden: rh = gelu(pos @ Wr1 + br1)   [NT, Hr]
    gemm_wmma_f16<0><<<dim3(Hr / 64, NT / 128), 256, 0, stream>>>(
        pos16, wr1h, br1, nullptr, 0, rh16, nullptr, NT, Hr, D);

    // 4) router logits + sparse top-2 weights
    router_kernel<<<NT / 8, 256, 0, stream>>>(rh16, Wr2, br2, temp,
                                              logits, tokw, NT, Hr);
    capacity_kernel<<<1, 256, 0, stream>>>(logits, loss_out, NT, 1.0f / (float)B);

    // 5) experts: combined = sum_e w_e * (gelu(pos@We1_e + be1_e) @ We2_e + be2_e)
    hipMemsetAsync(comb, 0, (size_t)NT * D * 4, stream);
    for (int e = 0; e < E; ++e) {
        gemm_wmma_f16<0><<<dim3(He / 64, NT / 128), 256, 0, stream>>>(
            pos16, we1h + (size_t)e * D * He, be1 + (size_t)e * He,
            nullptr, 0, eh16, nullptr, NT, He, D);
        gemm_wmma_f16<2><<<dim3(D / 64, NT / 128), 256, 0, stream>>>(
            eh16, we2h + (size_t)e * He * D, be2 + (size_t)e * D,
            tokw + e, E, nullptr, comb, NT, D, He);
    }

    // 6) combiner MLP
    cvt(comb, comb16, (long)NT * D);
    gemm_wmma_f16<0><<<dim3(Hc / 64, NT / 128), 256, 0, stream>>>(
        comb16, wc1h, bc1, nullptr, 0, ch16, nullptr, NT, Hc, D);
    gemm_wmma_f16<1><<<dim3(D / 64, NT / 128), 256, 0, stream>>>(
        ch16, wc2h, bc2, nullptr, 0, nullptr, cbuf, NT, D, Hc);

    // 7) LayerNorm -> final output
    ln_kernel<<<NT, 256, 0, stream>>>(cbuf, lng, lnb, out, D);
}
